// DecoderRNN_84267258347560
// MI455X (gfx1250) — matrix operand order repaired and checked by reading, not compiled, not run on hardware
//
#include <hip/hip_runtime.h>
#include <hip/hip_bf16.h>

// ---------------- problem constants ----------------
constexpr int BB  = 128;    // batch
constexpr int NN  = 196;    // #feature locations
constexpr int ENC = 2048;   // encoder dim
constexpr int DEC = 512;    // decoder / att / emb dim
constexpr int ATT = 512;
constexpr int EMB = 512;
constexpr int VV  = 10000;  // vocab
constexpr int VVP = 10048;  // vocab padded to multiple of 64
constexpr int TT  = 21;     // caption length
constexpr int TS  = 20;     // time steps (T-1)

typedef __attribute__((ext_vector_type(16))) __bf16 v16bf;
typedef __attribute__((ext_vector_type(8)))  float  v8f;

// f32 -> bf16 bits, round-to-nearest-even
__device__ __forceinline__ unsigned short bf16bits(float f) {
    union { float f; unsigned u; } x; x.f = f;
    unsigned r = x.u + 0x7FFFu + ((x.u >> 16) & 1u);
    return (unsigned short)(r >> 16);
}
__device__ __forceinline__ __bf16 f2bf(float f) {
    unsigned short hs = bf16bits(f);
    union { unsigned short s; __bf16 b; } y; y.s = hs;
    return y.b;
}
__device__ __forceinline__ float sigm(float x) {
    return 1.0f / (1.0f + __expf(-x));
}

union BU { float4 f[2]; v16bf v; };

// ---------------------------------------------------------------------------
// Transpose + convert weights: B (K x N, f32) -> Bt (N x K, bf16 bits).
// ---------------------------------------------------------------------------
__global__ __launch_bounds__(256)
void transpose_bf16_kernel(const float* __restrict__ B,
                           unsigned short* __restrict__ Bt, int K, int N)
{
    __shared__ float tile[32][33];
    const int k0 = blockIdx.y * 32, n0 = blockIdx.x * 32;
    #pragma unroll
    for (int i = 0; i < 32; i += 8) {
        int k = k0 + threadIdx.y + i, n = n0 + threadIdx.x;
        tile[threadIdx.y + i][threadIdx.x] =
            (k < K && n < N) ? B[(long)k * N + n] : 0.0f;
    }
    __syncthreads();
    #pragma unroll
    for (int i = 0; i < 32; i += 8) {
        int n = n0 + threadIdx.y + i, k = k0 + threadIdx.x;
        if (n < N && k < K)
            Bt[(long)n * K + k] = bf16bits(tile[threadIdx.x][threadIdx.y + i]);
    }
}

__global__ void fill_u16_kernel(unsigned short* __restrict__ p, int n)
{
    int i = blockIdx.x * blockDim.x + threadIdx.x;
    if (i < n) p[i] = 0;
}

__global__ void cvt_f32_bf16_kernel(const float* __restrict__ s,
                                    unsigned short* __restrict__ d, int n)
{
    int i = blockIdx.x * blockDim.x + threadIdx.x;
    if (i < n) d[i] = bf16bits(s[i]);
}

// ---------------------------------------------------------------------------
// GEMM: C[M,N] = A[M,K] x Bt[N,K](bf16, row stride ldbt)
//       (+bias0[N]) (+bias1[N]) (optionally += existing C)
// A is bf16 [M,K] when ABF16, else f32 [M,K] (converted in-register).
// One wave computes a 16x64 strip. The k-loop is completely branch-free:
// the caller guarantees Bt has ceil(N/64)*64 valid (possibly zero-padded)
// rows, so only the stores carry a scalar edge guard. Tile indices are
// forced into SGPRs (readfirstlane) => no divergent control flow (WMMA
// requires EXEC == all-ones). K % 32 == 0, N % 16 == 0.
// ---------------------------------------------------------------------------
template <bool ABF16>
__global__ __launch_bounds__(256)
void gemm_wmma(const void* __restrict__ Aany,
               const unsigned short* __restrict__ Bt,
               float* __restrict__ C,
               const float* __restrict__ bias0, const float* __restrict__ bias1,
               int M, int N, int K, int ldbt, long ldc,
               int n_strips, int accumulate)
{
    const int wv_in_blk = __builtin_amdgcn_readfirstlane(threadIdx.x) >> 5;
    const int wave = blockIdx.x * (blockDim.x >> 5) + wv_in_blk;
    const int lane = threadIdx.x & 31;
    const int m_tile  = wave / n_strips;
    const int n_strip = wave - m_tile * n_strips;
    if (m_tile * 16 >= M) return;

    const int hi    = lane >> 4;          // 0/1 half-wave
    const int lo    = lane & 15;
    const int kgrpA = hi << 4;            // A: +16 for upper half
    const int col_base = n_strip * 64 + lo;

    v8f acc[4];
    const v8f vzero = {0.f,0.f,0.f,0.f,0.f,0.f,0.f,0.f};
    acc[0] = vzero; acc[1] = vzero; acc[2] = vzero; acc[3] = vzero;

    const float*          ArowF = (const float*)Aany          + (long)(m_tile * 16 + lo) * K;
    const unsigned short* ArowH = (const unsigned short*)Aany + (long)(m_tile * 16 + lo) * K;

    for (int k0 = 0; k0 < K; k0 += 32) {
        // A fragment (16x32): element e -> K = k0 + 16*hi + e
        v16bf afrag;
        if constexpr (ABF16) {
            BU bu;
            bu.f[0] = *(const float4*)(ArowH + k0 + kgrpA);
            bu.f[1] = *(const float4*)(ArowH + k0 + 8 + kgrpA);
            afrag = bu.v;
        } else {
            const float4* p0 = (const float4*)(ArowF + k0 + kgrpA);
            const float4* p1 = (const float4*)(ArowF + k0 + 8 + kgrpA);
            float4 x0 = p0[0], x1 = p0[1];
            float4 y0 = p1[0], y1 = p1[1];
            afrag[0]  = f2bf(x0.x); afrag[1]  = f2bf(x0.y);
            afrag[2]  = f2bf(x0.z); afrag[3]  = f2bf(x0.w);
            afrag[4]  = f2bf(x1.x); afrag[5]  = f2bf(x1.y);
            afrag[6]  = f2bf(x1.z); afrag[7]  = f2bf(x1.w);
            afrag[8]  = f2bf(y0.x); afrag[9]  = f2bf(y0.y);
            afrag[10] = f2bf(y0.z); afrag[11] = f2bf(y0.w);
            afrag[12] = f2bf(y1.x); afrag[13] = f2bf(y1.y);
            afrag[14] = f2bf(y1.z); afrag[15] = f2bf(y1.w);
        }
        // B fragment: element e -> K = koffB + e  (32 contiguous bytes/lane)
        const unsigned short* Bk = Bt + k0 + (hi << 4);
        #pragma unroll
        for (int j = 0; j < 4; ++j) {
            const float4* bp = (const float4*)(Bk + (long)(col_base + (j << 4)) * ldbt);
            BU bu;
            bu.f[0] = bp[0];
            bu.f[1] = bp[1];
            acc[j] = __builtin_amdgcn_wmma_f32_16x16x32_bf16(
                        false, afrag, false, bu.v, (short)0, acc[j], false, false);
        }
    }

    // store: VGPR r -> row = 16*m_tile + 8*hi + r, col = col_base + 16*j
    const int rbase = m_tile * 16 + (hi << 3);
    const int tile0 = n_strip * 64;                   // scalar
    #pragma unroll
    for (int j = 0; j < 4; ++j) {
        if (tile0 + (j << 4) < N) {                   // scalar edge guard
            const int col = col_base + (j << 4);
            float badd = 0.0f;
            if (bias0) badd += bias0[col];
            if (bias1) badd += bias1[col];
            #pragma unroll
            for (int r = 0; r < 8; ++r) {
                float* cp = C + (long)(rbase + r) * ldc + col;
                float v = acc[j][r] + badd;
                if (accumulate) v += *cp;
                *cp = v;
            }
        }
    }
}

// ---------------------------------------------------------------------------
// mean over N axis of features -> bf16 (GEMM A operand)
// ---------------------------------------------------------------------------
__global__ void mean_feat_kernel(const float* __restrict__ features,
                                 unsigned short* __restrict__ mean_bf)
{
    int idx = blockIdx.x * blockDim.x + threadIdx.x;
    if (idx >= BB * ENC) return;
    int b = idx / ENC, e = idx - b * ENC;
    const float* p = features + (long)b * NN * ENC + e;
    float s = 0.0f;
    for (int n = 0; n < NN; ++n) s += p[(long)n * ENC];
    mean_bf[idx] = bf16bits(s * (1.0f / (float)NN));
}

__global__ void gather_emb_kernel(const float* __restrict__ emb,
                                  const int* __restrict__ captions,
                                  int t, unsigned short* __restrict__ emb_bf)
{
    int idx = blockIdx.x * blockDim.x + threadIdx.x;
    if (idx >= BB * EMB) return;
    int b = idx / EMB, e = idx - b * EMB;
    int tok = captions[b * TT + t];
    emb_bf[idx] = bf16bits(emb[(long)tok * EMB + e]);
}

// ---------------------------------------------------------------------------
// Fused attention (one batch row per block): scores -> softmax -> context.
// Context is emitted as bf16 (GEMM A operand).
// ---------------------------------------------------------------------------
__global__ __launch_bounds__(256)
void attention_kernel(const float* __restrict__ u_hs,
                      const float* __restrict__ w_ah,
                      const float* __restrict__ A_w,
                      const float* __restrict__ A_b,
                      const float* __restrict__ features,
                      unsigned short* __restrict__ ctx_bf,
                      float* __restrict__ alphas, int t)
{
    __shared__ float s_wah[ATT];
    __shared__ float s_sc[256];
    __shared__ float s_red[256];

    const int b    = blockIdx.x;
    const int tid  = threadIdx.x;
    const int wv   = tid >> 5;
    const int lane = tid & 31;

    for (int a = tid; a < ATT; a += 256) s_wah[a] = w_ah[b * ATT + a];
    __syncthreads();

    for (int n = wv; n < NN; n += 8) {
        const float* u = u_hs + ((long)b * NN + n) * ATT;
        float p = 0.0f;
        for (int a = lane; a < ATT; a += 32)
            p += tanhf(u[a] + s_wah[a]) * A_w[a];
        #pragma unroll
        for (int off = 16; off > 0; off >>= 1)
            p += __shfl_xor(p, off, 32);
        if (lane == 0) s_sc[n] = p + A_b[0];
    }
    __syncthreads();

    // softmax over NN
    float v = (tid < NN) ? s_sc[tid] : -1e30f;
    s_red[tid] = v; __syncthreads();
    for (int s = 128; s > 0; s >>= 1) {
        if (tid < s) s_red[tid] = fmaxf(s_red[tid], s_red[tid + s]);
        __syncthreads();
    }
    const float mx = s_red[0];
    __syncthreads();
    float e = (tid < NN) ? __expf(s_sc[tid] - mx) : 0.0f;
    s_red[tid] = e; __syncthreads();
    for (int s = 128; s > 0; s >>= 1) {
        if (tid < s) s_red[tid] += s_red[tid + s];
        __syncthreads();
    }
    const float inv = 1.0f / s_red[0];
    __syncthreads();
    if (tid < NN) {
        float a = e * inv;
        s_sc[tid] = a;
        alphas[((long)b * TS + t) * NN + tid] = a;
    }
    __syncthreads();

    // context[b,:] = sum_n alpha[n] * features[b,n,:]  (float4 vectorized)
    const float* fb = features + (long)b * NN * ENC;
    for (int e4 = tid * 4; e4 < ENC; e4 += 1024) {
        const float4* fp = (const float4*)(fb + e4);
        float4 sa = {0.f, 0.f, 0.f, 0.f};
        for (int n = 0; n < NN; ++n) {
            float4 f = fp[n * (ENC / 4)];
            float a = s_sc[n];
            sa.x += a * f.x; sa.y += a * f.y;
            sa.z += a * f.z; sa.w += a * f.w;
        }
        ushort4 u;
        u.x = bf16bits(sa.x); u.y = bf16bits(sa.y);
        u.z = bf16bits(sa.z); u.w = bf16bits(sa.w);
        *(ushort4*)(ctx_bf + (long)b * ENC + e4) = u;
    }
}

// ---------------------------------------------------------------------------
// LSTM pointwise: consumes gates + c, produces new c (f32) and h (bf16).
// ---------------------------------------------------------------------------
__global__ void lstm_pointwise(const float* __restrict__ gates,
                               unsigned short* __restrict__ h_bf,
                               float* __restrict__ c)
{
    int idx = blockIdx.x * blockDim.x + threadIdx.x;
    if (idx >= BB * DEC) return;
    int b = idx / DEC, d = idx - b * DEC;
    const float* g = gates + (long)b * 4 * DEC;
    float gi = g[d], gf = g[DEC + d], gg = g[2 * DEC + d], go = g[3 * DEC + d];
    float cn = sigm(gf) * c[idx] + sigm(gi) * tanhf(gg);
    float hn = sigm(go) * tanhf(cn);
    c[idx] = cn;
    h_bf[idx] = bf16bits(hn);
}

// ---------------------------------------------------------------------------
template <bool ABF16>
static void launch_gemm(const void* A, const unsigned short* Bt, float* C,
                        const float* b0, const float* b1,
                        int M, int N, int K, int ldbt, long ldc,
                        int acc, hipStream_t s)
{
    int n_strips = (N + 63) / 64;
    int m_tiles  = (M + 15) / 16;
    int waves    = m_tiles * n_strips;
    int blocks   = (waves + 7) / 8;
    gemm_wmma<ABF16><<<blocks, 256, 0, s>>>(A, Bt, C, b0, b1,
                                            M, N, K, ldbt, ldc, n_strips, acc);
}

static void launch_transpose(const float* B, unsigned short* Bt,
                             int K, int N, hipStream_t s)
{
    dim3 blk(32, 8);
    dim3 grd((N + 31) / 32, (K + 31) / 32);
    transpose_bf16_kernel<<<grd, blk, 0, s>>>(B, Bt, K, N);
}

extern "C" void kernel_launch(void* const* d_in, const int* in_sizes, int n_in,
                              void* d_out, int out_size, void* d_ws, size_t ws_size,
                              hipStream_t stream)
{
    const float* features = (const float*)d_in[0];
    const int*   captions = (const int*)  d_in[1];
    const float* emb      = (const float*)d_in[2];
    const float* U_w      = (const float*)d_in[3];
    const float* U_b      = (const float*)d_in[4];
    const float* W_w      = (const float*)d_in[5];
    const float* W_b      = (const float*)d_in[6];
    const float* A_w      = (const float*)d_in[7];
    const float* A_b      = (const float*)d_in[8];
    const float* ih_w     = (const float*)d_in[9];
    const float* ih_b     = (const float*)d_in[10];
    const float* ic_w     = (const float*)d_in[11];
    const float* ic_b     = (const float*)d_in[12];
    const float* lstm_Wih = (const float*)d_in[13];
    const float* lstm_Whh = (const float*)d_in[14];
    const float* lstm_bih = (const float*)d_in[15];
    const float* lstm_bhh = (const float*)d_in[16];
    const float* fcn_w    = (const float*)d_in[17];
    const float* fcn_b    = (const float*)d_in[18];

    float* preds  = (float*)d_out;                       // (B, TS, V)
    float* alphas = (float*)d_out + (long)BB * TS * VV;  // (B, TS, NN)

    // ---- workspace: f32 scratch, then bf16 activations + weight copies ----
    float* wsf   = (float*)d_ws;
    float* u_hs  = wsf;  wsf += (long)BB * NN * ATT;   // 12.8M
    float* h0f   = wsf;  wsf += (long)BB * DEC;
    float* c     = wsf;  wsf += (long)BB * DEC;
    float* w_ah  = wsf;  wsf += (long)BB * ATT;
    float* gates = wsf;  wsf += (long)BB * 4 * DEC;

    unsigned short* wsh = (unsigned short*)wsf;
    unsigned short* mean_bf = wsh;  wsh += (long)BB * ENC;
    unsigned short* emb_bf  = wsh;  wsh += (long)BB * EMB;
    unsigned short* ctx_bf  = wsh;  wsh += (long)BB * ENC;
    unsigned short* h_bf    = wsh;  wsh += (long)BB * DEC;
    unsigned short* U_wT    = wsh;  wsh += (long)ATT * ENC;               // 512x2048
    unsigned short* ih_wT   = wsh;  wsh += (long)DEC * ENC;               // 512x2048
    unsigned short* ic_wT   = wsh;  wsh += (long)DEC * ENC;               // 512x2048
    unsigned short* W_wT    = wsh;  wsh += (long)ATT * DEC;               // 512x512
    unsigned short* WihT    = wsh;  wsh += (long)(4 * DEC) * (EMB + ENC); // 2048x2560
    unsigned short* WhhT    = wsh;  wsh += (long)(4 * DEC) * DEC;         // 2048x512
    unsigned short* fcnT    = wsh;  wsh += (long)VVP * DEC;               // 10048x512 (padded)

    // ---- prologue ----
    launch_transpose(U_w,      U_wT,  ENC, ATT,           stream);
    launch_transpose(ih_w,     ih_wT, ENC, DEC,           stream);
    launch_transpose(ic_w,     ic_wT, ENC, DEC,           stream);
    launch_transpose(W_w,      W_wT,  DEC, ATT,           stream);
    launch_transpose(lstm_Wih, WihT,  EMB + ENC, 4 * DEC, stream);
    launch_transpose(lstm_Whh, WhhT,  DEC, 4 * DEC,       stream);
    launch_transpose(fcn_w,    fcnT,  DEC, VV,            stream);
    // zero the pad rows of fcnT so edge-strip loads are benign
    {
        int padn = (VVP - VV) * DEC;
        fill_u16_kernel<<<(padn + 255) / 256, 256, 0, stream>>>(fcnT + (long)VV * DEC, padn);
    }

    mean_feat_kernel<<<(BB * ENC + 255) / 256, 256, 0, stream>>>(features, mean_bf);

    // h0 / c0  (h0 -> f32 -> bf16 state; c stays f32)
    launch_gemm<true>(mean_bf, ih_wT, h0f, ih_b, nullptr, BB, DEC, ENC, ENC, DEC, 0, stream);
    launch_gemm<true>(mean_bf, ic_wT, c,   ic_b, nullptr, BB, DEC, ENC, ENC, DEC, 0, stream);
    cvt_f32_bf16_kernel<<<(BB * DEC + 255) / 256, 256, 0, stream>>>(h0f, h_bf, BB * DEC);

    // u_hs = features @ U_w + U_b   (25088 x 512, K = 2048; f32 A path)
    launch_gemm<false>(features, U_wT, u_hs, U_b, nullptr,
                       BB * NN, ATT, ENC, ENC, ATT, 0, stream);

    // ---- sequential decode ----
    for (int t = 0; t < TS; ++t) {
        gather_emb_kernel<<<(BB * EMB + 255) / 256, 256, 0, stream>>>(emb, captions, t, emb_bf);

        // w_ah = h @ W_w + W_b
        launch_gemm<true>(h_bf, W_wT, w_ah, W_b, nullptr, BB, ATT, DEC, DEC, ATT, 0, stream);

        attention_kernel<<<BB, 256, 0, stream>>>(u_hs, w_ah, A_w, A_b,
                                                 features, ctx_bf, alphas, t);

        // gates = emb_t @ Wih[0:512,:] + bih + bhh
        launch_gemm<true>(emb_bf, WihT, gates, lstm_bih, lstm_bhh,
                          BB, 4 * DEC, EMB, EMB + ENC, 4 * DEC, 0, stream);
        // gates += context @ Wih[512:2560,:]
        launch_gemm<true>(ctx_bf, WihT + EMB, gates, nullptr, nullptr,
                          BB, 4 * DEC, ENC, EMB + ENC, 4 * DEC, 1, stream);
        // gates += h @ Whh
        launch_gemm<true>(h_bf, WhhT, gates, nullptr, nullptr,
                          BB, 4 * DEC, DEC, DEC, 4 * DEC, 1, stream);

        lstm_pointwise<<<(BB * DEC + 255) / 256, 256, 0, stream>>>(gates, h_bf, c);

        // preds[:, t, :] = h @ fcn_w + fcn_b  (strided output rows)
        launch_gemm<true>(h_bf, fcnT, preds + (long)t * VV, fcn_b, nullptr,
                          BB, VV, DEC, DEC, (long)TS * VV, 0, stream);
    }
    (void)in_sizes; (void)n_in; (void)out_size; (void)ws_size;
}
